// DifferentialSelfAttention_80075370266692
// MI455X (gfx1250) — matrix operand order repaired and checked by reading, not compiled, not run on hardware
//
#include <hip/hip_runtime.h>
#include <hip/hip_bf16.h>

// ---------------------------------------------------------------------------
// Differential self-attention for gfx1250 (MI455X), f16 WMMA + f32 accumulate
// ---------------------------------------------------------------------------

#define S_LEN   2048
#define BATCH   2
#define NH      16
#define HDIM    64
#define DMODEL  1024
#define MROWS   (BATCH * S_LEN)     // 4096
#define LAMBDA0 0.2f                // 0.8 - 0.6*exp(0)

typedef __attribute__((ext_vector_type(16))) _Float16 v16h;
typedef __attribute__((ext_vector_type(8)))  _Float16 v8h;
typedef __attribute__((ext_vector_type(8)))  float    v8f;

union F16x16 { v16h v; v8h h[2]; };

// D = A(16x32) * B(32x16) + C, f16 in / f32 out
__device__ __forceinline__ v8f wmma16(v16h a, v16h b, v8f c) {
  return __builtin_amdgcn_wmma_f32_16x16x32_f16(false, a, false, b,
                                                (short)0, c, false, false);
}

// A-matrix fragment (16x32, row-major in memory, row stride = ld halves).
// ISA layout: lanes 0-15 hold K=0..7 (v0-3) and 16..23 (v4-7) of row M=lane;
// lanes 16-31 hold K=8..15 and 24..31 of row M=lane-16.
__device__ __forceinline__ v16h load_frag_A(const _Float16* base, int ld) {
  int lane = threadIdx.x & 31;
  int row  = lane & 15;
  int koff = (lane >> 4) << 3;
  F16x16 f;
  f.h[0] = *(const v8h*)(base + (size_t)row * ld + koff);
  f.h[1] = *(const v8h*)(base + (size_t)row * ld + 16 + koff);
  return f.v;
}

// B-matrix fragment (32x16). Memory holds B^T row-major: row n = output
// column n, row stride = ld halves. ISA layout: lanes 0-15 hold K=0..7 and
// 16..23 of column N=lane; lanes 16-31 hold K=8..15 and 24..31 of column N=lane-16.
__device__ __forceinline__ v16h load_frag_B(const _Float16* base, int ld) {
  int lane = threadIdx.x & 31;
  int col  = lane & 15;
  int koff = (lane >> 4) << 3;
  F16x16 f;
  f.h[0] = *(const v8h*)(base + (size_t)col * ld + koff);
  f.h[1] = *(const v8h*)(base + (size_t)col * ld + 16 + koff);
  return f.v;
}

// 16-lane butterfly reductions via ds_swizzle (group-of-32 mode, and=0x1f,
// xor masks 1/2/4/8 stay inside each 16-lane half -> per-row reductions of a
// WMMA C-layout tile). No address VGPR needed, unlike ds_bpermute.
__device__ __forceinline__ float swz(float v, int pat);
#define SWZ(v, pat) \
  __builtin_bit_cast(float, __builtin_amdgcn_ds_swizzle(__builtin_bit_cast(int, (v)), (pat)))

__device__ __forceinline__ float rmax16(float v) {
  v = fmaxf(v, SWZ(v, 0x041F));
  v = fmaxf(v, SWZ(v, 0x081F));
  v = fmaxf(v, SWZ(v, 0x101F));
  v = fmaxf(v, SWZ(v, 0x201F));
  return v;
}
__device__ __forceinline__ float rsum16(float v) {
  v += SWZ(v, 0x041F);
  v += SWZ(v, 0x081F);
  v += SWZ(v, 0x101F);
  v += SWZ(v, 0x201F);
  return v;
}

// ---------------------------------------------------------------------------
// f32 -> f16 convert
// ---------------------------------------------------------------------------
__global__ __launch_bounds__(256) void cvt_f16_kernel(const float* __restrict__ src,
                                                      _Float16* __restrict__ dst, int n) {
  int i = blockIdx.x * blockDim.x + threadIdx.x;
  int stride = gridDim.x * blockDim.x;
  for (; i < n; i += stride) dst[i] = (_Float16)src[i];
}

// ---------------------------------------------------------------------------
// C[M,N] = A[M,K](f16) @ W[N,K]^T(f16) + bias, 4 waves/block, wave = 16x64 tile
// ---------------------------------------------------------------------------
template <bool OUT_F16>
__global__ __launch_bounds__(128, 1) void gemm_wmma_kernel(
    const _Float16* __restrict__ A, const _Float16* __restrict__ W,
    const float* __restrict__ bias, void* __restrict__ C, int M, int N, int K) {
  int wave = threadIdx.x >> 5;
  int lane = threadIdx.x & 31;
  int m0 = blockIdx.y * 64 + wave * 16;
  int n0 = blockIdx.x * 64;

  v8f acc[4] = {};
  for (int kb = 0; kb < K; kb += 32) {
    v16h a = load_frag_A(A + (size_t)m0 * K + kb, K);
#pragma unroll
    for (int nt = 0; nt < 4; ++nt) {
      v16h b = load_frag_B(W + (size_t)(n0 + nt * 16) * K + kb, K);
      acc[nt] = wmma16(a, b, acc[nt]);
    }
  }

  int rowoff = (lane >> 4) << 3;
  int col_l  = lane & 15;
#pragma unroll
  for (int nt = 0; nt < 4; ++nt) {
    int col = n0 + nt * 16 + col_l;
    float bv = bias[col];
#pragma unroll
    for (int r = 0; r < 8; ++r) {
      size_t idx = (size_t)(m0 + rowoff + r) * N + col;
      float val = acc[nt][r] + bv;
      if (OUT_F16) ((_Float16*)C)[idx] = (_Float16)val;
      else         ((float*)C)[idx] = val;
    }
  }
}

// ---------------------------------------------------------------------------
// RoPE + repack: q/k [B,S,2D] -> q1/q2/k1/k2 [B,H,S,64] (q scaled by 1/8),
// v [B,S,D] -> vt [B,H,64,S]
// ---------------------------------------------------------------------------
__global__ __launch_bounds__(64) void rope_pack_kernel(
    const _Float16* __restrict__ qh, const _Float16* __restrict__ kh,
    const _Float16* __restrict__ vh,
    _Float16* __restrict__ q1, _Float16* __restrict__ q2,
    _Float16* __restrict__ k1, _Float16* __restrict__ k2,
    _Float16* __restrict__ vt) {
  int idx = blockIdx.x;
  int s = idx & (S_LEN - 1);
  int h = (idx >> 11) & (NH - 1);
  int b = idx >> 15;
  int d  = threadIdx.x;      // 0..63
  int dh = d & 31;

  // inv_freq = 10000^(-dh/32) ;  ln(10000) = 9.210340371976184
  float invf = __expf(-(float)dh * (9.210340371976184f / 32.0f));
  float ang = (float)s * invf;
  float c = cosf(ang), sn = sinf(ang);
  bool lo = (d < 32);

  size_t rowqk = ((size_t)b * S_LEN + s) * (2 * DMODEL) + (size_t)h * (2 * HDIM);
  size_t outb  = (((size_t)(b * NH + h)) * S_LEN + s) * HDIM + d;

  {
    float x1 = (float)qh[rowqk + dh], x2 = (float)qh[rowqk + dh + 32];
    float o = lo ? (x1 * c - x2 * sn) : (x1 * sn + x2 * c);
    q1[outb] = (_Float16)(o * 0.125f);
  }
  {
    float x1 = (float)qh[rowqk + 64 + dh], x2 = (float)qh[rowqk + 64 + dh + 32];
    float o = lo ? (x1 * c - x2 * sn) : (x1 * sn + x2 * c);
    q2[outb] = (_Float16)(o * 0.125f);
  }
  {
    float x1 = (float)kh[rowqk + dh], x2 = (float)kh[rowqk + dh + 32];
    float o = lo ? (x1 * c - x2 * sn) : (x1 * sn + x2 * c);
    k1[outb] = (_Float16)o;
  }
  {
    float x1 = (float)kh[rowqk + 64 + dh], x2 = (float)kh[rowqk + 64 + dh + 32];
    float o = lo ? (x1 * c - x2 * sn) : (x1 * sn + x2 * c);
    k2[outb] = (_Float16)o;
  }
  size_t rowv = ((size_t)b * S_LEN + s) * DMODEL + (size_t)h * HDIM + d;
  vt[(((size_t)(b * NH + h)) * HDIM + d) * S_LEN + s] = vh[rowv];
}

// ---------------------------------------------------------------------------
// lambda per head
// ---------------------------------------------------------------------------
__global__ __launch_bounds__(32) void lambda_kernel(
    const float* __restrict__ lq1, const float* __restrict__ lk1,
    const float* __restrict__ lq2, const float* __restrict__ lk2,
    float* __restrict__ lam) {
  int h = threadIdx.x;
  if (h < NH) {
    float d1 = 0.f, d2 = 0.f;
    for (int i = 0; i < HDIM; ++i) {
      d1 += lq1[h * HDIM + i] * lk1[h * HDIM + i];
      d2 += lq2[h * HDIM + i] * lk2[h * HDIM + i];
    }
    lam[h] = LAMBDA0 + __expf(d1) - __expf(d2);
  }
}

// ---------------------------------------------------------------------------
// One score set: s = Q.K^T (masked), online-softmax update of (m,l), rescale
// acc, probabilities -> LDS (16x32 f16 row-major).
// ---------------------------------------------------------------------------
__device__ __forceinline__ void score_softmax(const v16h* aq,
                                              const _Float16* __restrict__ khb,
                                              int kb, int qbase,
                                              _Float16* Plds,
                                              v8f* acc, float* m, float* l) {
  int lane = threadIdx.x & 31;

  v8f s[2];
#pragma unroll
  for (int nt = 0; nt < 2; ++nt) {
    const _Float16* kp = khb + (size_t)(kb + nt * 16) * HDIM;
    v16h b0 = load_frag_B(kp + 0,  HDIM);
    v16h b1 = load_frag_B(kp + 32, HDIM);
    v8f sc = {};
    sc = wmma16(aq[0], b0, sc);
    sc = wmma16(aq[1], b1, sc);
    s[nt] = sc;
  }

  // causal mask (row i = qbase + r + 8*(lane>=16); col j = kb + nt*16 + lane%16)
  int qrow0 = qbase + ((lane >> 4) << 3);
  if (kb + 31 > qbase) {
#pragma unroll
    for (int nt = 0; nt < 2; ++nt) {
      int j = kb + nt * 16 + (lane & 15);
#pragma unroll
      for (int r = 0; r < 8; ++r)
        if (j > qrow0 + r) s[nt][r] = -1.0e30f;
    }
  }

  float mnew[8], f[8], rs[8];
#pragma unroll
  for (int r = 0; r < 8; ++r) {
    float t = rmax16(fmaxf(s[0][r], s[1][r]));
    mnew[r] = fmaxf(m[r], t);
    f[r] = __expf(m[r] - mnew[r]);
    rs[r] = 0.f;
  }

  int lcol = lane & 15;
  int lrow = (lane >> 4) << 3;
#pragma unroll
  for (int nt = 0; nt < 2; ++nt) {
#pragma unroll
    for (int r = 0; r < 8; ++r) {
      float p = __expf(s[nt][r] - mnew[r]);
      rs[r] += p;
      Plds[(size_t)(r + lrow) * 32 + nt * 16 + lcol] = (_Float16)p;
    }
  }
#pragma unroll
  for (int r = 0; r < 8; ++r) {
    float t = rsum16(rs[r]);
    l[r] = l[r] * f[r] + t;
    m[r] = mnew[r];
  }
#pragma unroll
  for (int nt = 0; nt < 4; ++nt)
#pragma unroll
    for (int r = 0; r < 8; ++r) acc[nt][r] *= f[r];
}

// ---------------------------------------------------------------------------
// Differential flash attention. grid = (S/64, B*H), block = 128 (4 waves).
// Each wave owns 16 query rows. Both score sets share the V fragments.
// Output: O [B, S, H*64] f16.
// ---------------------------------------------------------------------------
__global__ __launch_bounds__(128, 1) void diff_attn_kernel(
    const _Float16* __restrict__ q1h, const _Float16* __restrict__ q2h,
    const _Float16* __restrict__ k1h, const _Float16* __restrict__ k2h,
    const _Float16* __restrict__ vt, const float* __restrict__ lam,
    const float* __restrict__ hnw, _Float16* __restrict__ O) {
  __shared__ _Float16 Pst[4][2][16 * 32];

  int bh = blockIdx.y;
  int h = bh & (NH - 1);
  int b = bh >> 4;
  int wave = threadIdx.x >> 5;
  int lane = threadIdx.x & 31;
  int qbase = blockIdx.x * 64 + wave * 16;

  size_t headq = ((size_t)bh * S_LEN + qbase) * HDIM;
  const _Float16* k1b = k1h + (size_t)bh * S_LEN * HDIM;
  const _Float16* k2b = k2h + (size_t)bh * S_LEN * HDIM;
  const _Float16* vtb = vt + (size_t)bh * HDIM * S_LEN;
  _Float16* P1 = Pst[wave][0];
  _Float16* P2 = Pst[wave][1];

  v16h a1[2], a2[2];
  a1[0] = load_frag_A(q1h + headq, HDIM);
  a1[1] = load_frag_A(q1h + headq + 32, HDIM);
  a2[0] = load_frag_A(q2h + headq, HDIM);
  a2[1] = load_frag_A(q2h + headq + 32, HDIM);

  v8f acc1[4] = {}, acc2[4] = {};
  float m1[8], l1[8], m2[8], l2[8];
#pragma unroll
  for (int r = 0; r < 8; ++r) {
    m1[r] = -3.0e38f; m2[r] = -3.0e38f;
    l1[r] = 0.f;      l2[r] = 0.f;
  }

  int kend = qbase + 16;                 // keys 0 .. qbase+15 can be valid
  for (int kb = 0; kb < kend; kb += 32) {
    score_softmax(a1, k1b, kb, qbase, P1, acc1, m1, l1);
    score_softmax(a2, k2b, kb, qbase, P2, acc2, m2, l2);

    // P tiles as A-fragments from LDS; V fragments shared by both sets
    asm volatile("s_wait_dscnt 0" ::: "memory");
    F16x16 pa1, pa2;
    {
      int row  = lane & 15;
      int koff = (lane >> 4) << 3;
      pa1.h[0] = *(const v8h*)(P1 + (size_t)row * 32 + koff);
      pa1.h[1] = *(const v8h*)(P1 + (size_t)row * 32 + 16 + koff);
      pa2.h[0] = *(const v8h*)(P2 + (size_t)row * 32 + koff);
      pa2.h[1] = *(const v8h*)(P2 + (size_t)row * 32 + 16 + koff);
    }
#pragma unroll
    for (int nt = 0; nt < 4; ++nt) {
      v16h bv = load_frag_B(vtb + (size_t)(nt * 16) * S_LEN + kb, S_LEN);
      acc1[nt] = wmma16(pa1.v, bv, acc1[nt]);
      acc2[nt] = wmma16(pa2.v, bv, acc2[nt]);
    }
  }

  // ---- epilogue: a1/l1 - lam*a2/l2, RMSNorm over hd, head scale, (1-lam0)
  float lamh = lam[h];
  float ss[8];
#pragma unroll
  for (int r = 0; r < 8; ++r) ss[r] = 0.f;
#pragma unroll
  for (int nt = 0; nt < 4; ++nt) {
#pragma unroll
    for (int r = 0; r < 8; ++r) {
      float o = acc1[nt][r] / l1[r] - lamh * (acc2[nt][r] / l2[r]);
      acc1[nt][r] = o;                  // reuse acc1 as output storage
      ss[r] += o * o;
    }
  }
  float rinv[8];
#pragma unroll
  for (int r = 0; r < 8; ++r) {
    float t = rsum16(ss[r]);
    rinv[r] = rsqrtf(t * (1.0f / HDIM) + 1e-6f);
  }
  int lcol = lane & 15;
  int lrow = (lane >> 4) << 3;
#pragma unroll
  for (int nt = 0; nt < 4; ++nt) {
    float w = hnw[h * HDIM + nt * 16 + lcol] * (1.0f - LAMBDA0);
#pragma unroll
    for (int r = 0; r < 8; ++r) {
      float o = acc1[nt][r] * rinv[r] * w;
      size_t idx = ((size_t)b * S_LEN + qbase + lrow + r) * DMODEL +
                   (size_t)h * HDIM + nt * 16 + lcol;
      O[idx] = (_Float16)o;
    }
  }
}

// ---------------------------------------------------------------------------
// Host-side launcher
// ---------------------------------------------------------------------------
extern "C" void kernel_launch(void* const* d_in, const int* in_sizes, int n_in,
                              void* d_out, int out_size, void* d_ws, size_t ws_size,
                              hipStream_t stream) {
  const float* x   = (const float*)d_in[0];
  const float* Wq  = (const float*)d_in[1];
  const float* bq  = (const float*)d_in[2];
  const float* Wk  = (const float*)d_in[3];
  const float* bk  = (const float*)d_in[4];
  const float* Wv  = (const float*)d_in[5];
  const float* bv  = (const float*)d_in[6];
  const float* Wo  = (const float*)d_in[7];
  const float* bo  = (const float*)d_in[8];
  const float* hnw = (const float*)d_in[9];
  const float* lq1 = (const float*)d_in[10];
  const float* lk1 = (const float*)d_in[11];
  const float* lq2 = (const float*)d_in[12];
  const float* lk2 = (const float*)d_in[13];
  float* out = (float*)d_out;

  // ---- scratch layout (halves)
  _Float16* p = (_Float16*)d_ws;
  _Float16* xh  = p; p += (size_t)MROWS * DMODEL;          // 4M
  _Float16* Wqh = p; p += (size_t)2 * DMODEL * DMODEL;     // 2M
  _Float16* Wkh = p; p += (size_t)2 * DMODEL * DMODEL;     // 2M
  _Float16* Wvh = p; p += (size_t)DMODEL * DMODEL;         // 1M
  _Float16* Woh = p; p += (size_t)DMODEL * DMODEL;         // 1M
  _Float16* qhb = p; p += (size_t)MROWS * 2 * DMODEL;      // 8M
  _Float16* khb = p; p += (size_t)MROWS * 2 * DMODEL;      // 8M
  _Float16* vhb = p; p += (size_t)MROWS * DMODEL;          // 4M
  _Float16* q1h = p; p += (size_t)BATCH * NH * S_LEN * HDIM; // 4M
  _Float16* q2h = p; p += (size_t)BATCH * NH * S_LEN * HDIM;
  _Float16* k1h = p; p += (size_t)BATCH * NH * S_LEN * HDIM;
  _Float16* k2h = p; p += (size_t)BATCH * NH * S_LEN * HDIM;
  float* lamf = (float*)p;
  _Float16* vtb  = xh;   // alias: xh dead after projection GEMMs
  _Float16* Obuf = qhb;  // alias: qhb dead after rope_pack

  // ---- 1) f32 -> f16 converts
  {
    int n0 = MROWS * DMODEL;
    cvt_f16_kernel<<<(n0 + 255) / 256, 256, 0, stream>>>(x, xh, n0);
    int nw = 2 * DMODEL * DMODEL;
    cvt_f16_kernel<<<(nw + 255) / 256, 256, 0, stream>>>(Wq, Wqh, nw);
    cvt_f16_kernel<<<(nw + 255) / 256, 256, 0, stream>>>(Wk, Wkh, nw);
    int nv = DMODEL * DMODEL;
    cvt_f16_kernel<<<(nv + 255) / 256, 256, 0, stream>>>(Wv, Wvh, nv);
    cvt_f16_kernel<<<(nv + 255) / 256, 256, 0, stream>>>(Wo, Woh, nv);
  }

  // ---- 2) projection GEMMs (f16 out)
  {
    dim3 blk(128);
    gemm_wmma_kernel<true><<<dim3((2 * DMODEL) / 64, MROWS / 64), blk, 0, stream>>>(
        xh, Wqh, bq, qhb, MROWS, 2 * DMODEL, DMODEL);
    gemm_wmma_kernel<true><<<dim3((2 * DMODEL) / 64, MROWS / 64), blk, 0, stream>>>(
        xh, Wkh, bk, khb, MROWS, 2 * DMODEL, DMODEL);
    gemm_wmma_kernel<true><<<dim3(DMODEL / 64, MROWS / 64), blk, 0, stream>>>(
        xh, Wvh, bv, vhb, MROWS, DMODEL, DMODEL);
  }

  // ---- 3) lambda per head
  lambda_kernel<<<1, 32, 0, stream>>>(lq1, lk1, lq2, lk2, lamf);

  // ---- 4) RoPE + repack (+ V transpose into aliased xh)
  rope_pack_kernel<<<BATCH * NH * S_LEN, HDIM, 0, stream>>>(
      qhb, khb, vhb, q1h, q2h, k1h, k2h, vtb);

  // ---- 5) differential flash attention
  diff_attn_kernel<<<dim3(S_LEN / 64, BATCH * NH), 128, 0, stream>>>(
      q1h, q2h, k1h, k2h, vtb, lamf, hnw, Obuf);

  // ---- 6) output projection (f32 out)
  gemm_wmma_kernel<false><<<dim3(DMODEL / 64, MROWS / 64), 128, 0, stream>>>(
      Obuf, Woh, bo, out, MROWS, DMODEL, DMODEL);

  (void)in_sizes; (void)n_in; (void)out_size; (void)ws_size;
}
